// SwinTransformerTiny_35356170781029
// MI455X (gfx1250) — matrix operand order repaired and checked by reading, not compiled, not run on hardware
//
#include <hip/hip_runtime.h>
#include <hip/hip_bf16.h>

typedef __attribute__((ext_vector_type(16))) _Float16 v16h;
typedef __attribute__((ext_vector_type(8)))  _Float16 v8h;
typedef __attribute__((ext_vector_type(8)))  float    v8f;

#define WMMA_F32_F16(a, b, c) \
  __builtin_amdgcn_wmma_f32_16x16x32_f16(false, (a), false, (b), (short)0, (c), false, false)

__device__ inline v16h pack16(v8h lo, v8h hi) {
  union { v16h v; v8h p[2]; } u;
  u.p[0] = lo; u.p[1] = hi;
  return u.v;
}

__device__ inline float wave_sum(float v) {
#pragma unroll
  for (int o = 16; o > 0; o >>= 1) v += __shfl_xor(v, o, 32);
  return v;
}

__device__ inline float gelu_exact(float v) {
  return 0.5f * v * (1.f + erff(v * 0.70710678f));
}

// ---------------------------------------------------------------------------
// Patch embedding: conv 4x4 stride 4, NCHW input -> NHWC output (64,56,56,96)
// ---------------------------------------------------------------------------
__global__ void patch_embed_kernel(const float* __restrict__ x,
                                   const float* __restrict__ wgt,
                                   const float* __restrict__ bias,
                                   float* __restrict__ out) {
  size_t idx = (size_t)blockIdx.x * blockDim.x + threadIdx.x;
  const size_t total = 64ull * 56 * 56 * 96;
  if (idx >= total) return;
  int oc = (int)(idx % 96);
  size_t r = idx / 96;
  int ow = (int)(r % 56); r /= 56;
  int oh = (int)(r % 56);
  int b  = (int)(r / 56);
  float s = bias[oc];
#pragma unroll
  for (int ci = 0; ci < 3; ++ci)
#pragma unroll
    for (int kh = 0; kh < 4; ++kh)
#pragma unroll
      for (int kw = 0; kw < 4; ++kw)
        s += x[(((size_t)b * 3 + ci) * 224 + (oh * 4 + kh)) * 224 + (ow * 4 + kw)] *
             wgt[((oc * 3 + ci) * 4 + kh) * 4 + kw];
  out[idx] = s;
}

// ---------------------------------------------------------------------------
// Weight convert+transpose: W[N,K] f32 -> Wt[K,N] f16 (coalesced writes)
// ---------------------------------------------------------------------------
__global__ void wt_convert_kernel(const float* __restrict__ W, _Float16* __restrict__ Wt,
                                  int N, int K) {
  size_t idx = (size_t)blockIdx.x * blockDim.x + threadIdx.x;
  if (idx >= (size_t)N * K) return;
  int n = (int)(idx % N);
  int k = (int)(idx / N);
  Wt[idx] = (_Float16)W[(size_t)n * K + k];
}

// ---------------------------------------------------------------------------
// Generic LayerNorm, one wave per row; OutT selects f32 or f16 output.
// ---------------------------------------------------------------------------
template <typename OutT>
__global__ void ln_kernel(const float* __restrict__ in, const float* __restrict__ w,
                          const float* __restrict__ b, OutT* __restrict__ out,
                          int M, int C) {
  int wid  = (int)(((size_t)blockIdx.x * blockDim.x + threadIdx.x) >> 5);
  int lane = threadIdx.x & 31;
  if (wid >= M) return;
  const float* row = in + (size_t)wid * C;
  float s = 0.f;
  for (int c = lane; c < C; c += 32) s += row[c];
  float mu = wave_sum(s) / (float)C;
  float v = 0.f;
  for (int c = lane; c < C; c += 32) { float d = row[c] - mu; v += d * d; }
  float inv = rsqrtf(wave_sum(v) / (float)C + 1e-5f);
  OutT* o = out + (size_t)wid * C;
  for (int c = lane; c < C; c += 32) o[c] = (OutT)((row[c] - mu) * inv * w[c] + b[c]);
}

// ---------------------------------------------------------------------------
// LayerNorm + roll(-shift) + window partition: x (64,H,H,C) f32 -> xw f16
// ---------------------------------------------------------------------------
__global__ void ln_window_kernel(const float* __restrict__ x, const float* __restrict__ w,
                                 const float* __restrict__ b, _Float16* __restrict__ xw,
                                 int Bn, int H, int C, int shift) {
  int nh = H / 7;
  int row  = (int)(((size_t)blockIdx.x * blockDim.x + threadIdx.x) >> 5);
  int lane = threadIdx.x & 31;
  int nWin = nh * nh;
  int total = Bn * nWin * 49;
  if (row >= total) return;
  int t   = row % 49;
  int win = (row / 49) % nWin;
  int bb  = row / (49 * nWin);
  int wh = win / nh, ww = win % nh;
  int gh = wh * 7 + t / 7, gw = ww * 7 + t % 7;
  int oh = (gh + shift) % H, ow = (gw + shift) % H;
  const float* src = x + (((size_t)bb * H + oh) * H + ow) * C;
  float s = 0.f;
  for (int c = lane; c < C; c += 32) s += src[c];
  float mu = wave_sum(s) / (float)C;
  float v = 0.f;
  for (int c = lane; c < C; c += 32) { float d = src[c] - mu; v += d * d; }
  float inv = rsqrtf(wave_sum(v) / (float)C + 1e-5f);
  _Float16* o = xw + (size_t)row * C;
  for (int c = lane; c < C; c += 32) o[c] = (_Float16)((src[c] - mu) * inv * w[c] + b[c]);
}

// ---------------------------------------------------------------------------
// Reverse window partition + roll(+shift), residual add into x (f32 += f32)
// ---------------------------------------------------------------------------
__global__ void unwindow_add_kernel(float* __restrict__ x, const float* __restrict__ p,
                                    int Bn, int H, int C, int shift) {
  size_t idx = (size_t)blockIdx.x * blockDim.x + threadIdx.x;
  size_t total = (size_t)Bn * H * H * C;
  if (idx >= total) return;
  int c = (int)(idx % C);
  size_t r = idx / C;
  int w = (int)(r % H); r /= H;
  int h = (int)(r % H);
  int bb = (int)(r / H);
  int rh = (h - shift + H) % H, rw = (w - shift + H) % H;
  int nh = H / 7;
  int win = (rh / 7) * nh + (rw / 7);
  int t   = (rh % 7) * 7 + (rw % 7);
  size_t prow = ((size_t)bb * nh * nh + win) * 49 + t;
  x[idx] += p[prow * C + c];
}

// ---------------------------------------------------------------------------
// WMMA GEMM: C[M,N] = A[M,K](f16) @ Bt[K,N](f16) + bias (+resid) (+GELU)
// 32x32 tile per wave (2x2 accumulators, 4 WMMAs / 32-k-step), b128 loads.
// M % 32 == 0, N % 32 == 0, K % 32 == 0. Output to Cf (f32) or Ch (f16).
// ---------------------------------------------------------------------------
__global__ void __launch_bounds__(256)
gemm_wmma_kernel(const _Float16* __restrict__ A, const _Float16* __restrict__ Bt,
                 const float* __restrict__ bias, const float* __restrict__ resid,
                 float* __restrict__ Cf, _Float16* __restrict__ Ch,
                 int M, int N, int K, int act) {
  int lane = threadIdx.x & 31;
  int wave = threadIdx.x >> 5;
  int Mt = M >> 5, Nt = N >> 5;
  long tile = (long)blockIdx.x * 8 + wave;
  if (tile >= (long)Mt * Nt) return;
  int mt = (int)(tile % Mt), nt = (int)(tile / Mt);
  v8f acc[2][2] = {{{}, {}}, {{}, {}}};
  int l15 = lane & 15;
  int kb  = (lane >> 4) * 8;
  const _Float16* Arow0 = A + (size_t)(mt * 32 + l15) * K;
  const _Float16* Arow1 = Arow0 + (size_t)16 * K;
  const _Float16* Brow  = Bt + (size_t)lane * N + nt * 32;
  for (int k0 = 0; k0 < K; k0 += 32) {
    v16h a0 = pack16(*(const v8h*)(Arow0 + k0 + kb), *(const v8h*)(Arow0 + k0 + 16 + kb));
    v16h a1 = pack16(*(const v8h*)(Arow1 + k0 + kb), *(const v8h*)(Arow1 + k0 + 16 + kb));
    const _Float16* bp = Bt + (size_t)k0 * N + nt * 32 + l15;
    v16h b0, b1;
#pragma unroll
    for (int j = 0; j < 16; ++j) {
      int kk = j < 8 ? kb + j : 8 + kb + j;
      b0[j] = bp[(size_t)kk * N];
      b1[j] = bp[(size_t)kk * N + 16];
    }
    acc[0][0] = WMMA_F32_F16(a0, b0, acc[0][0]);
    acc[0][1] = WMMA_F32_F16(a0, b1, acc[0][1]);
    acc[1][0] = WMMA_F32_F16(a1, b0, acc[1][0]);
    acc[1][1] = WMMA_F32_F16(a1, b1, acc[1][1]);
  }
  int mo = (lane >> 4) * 8;
#pragma unroll
  for (int sm = 0; sm < 2; ++sm)
#pragma unroll
    for (int sn = 0; sn < 2; ++sn) {
      int n = nt * 32 + sn * 16 + l15;
      float bb = bias ? bias[n] : 0.f;
#pragma unroll
      for (int r = 0; r < 8; ++r) {
        int m = mt * 32 + sm * 16 + mo + r;
        float v = acc[sm][sn][r] + bb;
        if (resid) v += resid[(size_t)m * N + n];
        if (act) v = gelu_exact(v);
        if (Ch) Ch[(size_t)m * N + n] = (_Float16)v;
        else    Cf[(size_t)m * N + n] = v;
      }
    }
}

// ---------------------------------------------------------------------------
// Window attention: one block (128 threads = 4 waves) per (window, head).
// head_dim = 32. q/k/v staged in f16 LDS; S = qk^T (WMMA, f32 acc);
// softmax (f32, q-scale folded in) + rel-pos bias + shift mask; O = P@V (WMMA).
// qkv: [nWin*49, 3C] f16, out: [nWin*49, C] f16.
// ---------------------------------------------------------------------------
__global__ void __launch_bounds__(128)
attn_kernel(const _Float16* __restrict__ qkv, const float* __restrict__ rpb,
            _Float16* __restrict__ out, int nH, int C, int nh, int shift, int H) {
  __shared__ __align__(16) _Float16 qs[64 * 32];
  __shared__ __align__(16) _Float16 ks[64 * 32];
  __shared__ __align__(16) _Float16 vs[64 * 32];
  __shared__ float Ss[64 * 64];
  int bid  = blockIdx.x;
  int h    = bid % nH;
  int win  = bid / nH;
  int tid  = threadIdx.x;
  int lane = tid & 31;
  int wave = tid >> 5;
  int C3   = 3 * C;
  for (int i = 49 * 32 + tid; i < 64 * 32; i += 128) {
    qs[i] = (_Float16)0.f; ks[i] = (_Float16)0.f; vs[i] = (_Float16)0.f;
  }
  for (int i = tid; i < 49 * 32; i += 128) {
    int t = i >> 5, d = i & 31;
    const _Float16* base = qkv + ((size_t)win * 49 + t) * C3;
    qs[i] = base[h * 32 + d];
    ks[i] = base[(nH + h) * 32 + d];
    vs[i] = base[(2 * nH + h) * 32 + d];
  }
  __syncthreads();
  // ---- S = q @ k^T : wave owns m-tile `wave`, loops 4 n-tiles ----
  {
    int mt = wave;
    int am = mt * 16 + (lane & 15);
    int kb = (lane >> 4) * 8;
    v16h a = pack16(*(const v8h*)(qs + am * 32 + kb), *(const v8h*)(qs + am * 32 + 16 + kb));
    for (int nt2 = 0; nt2 < 4; ++nt2) {
      v16h b;
#pragma unroll
      for (int j = 0; j < 16; ++j) b[j] = ks[(nt2 * 16 + (lane & 15)) * 32 + (j < 8 ? kb + j : 8 + kb + j)];
      v8f acc = {};
      acc = WMMA_F32_F16(a, b, acc);
      int n  = nt2 * 16 + (lane & 15);
      int mo = (lane >> 4) * 8;
#pragma unroll
      for (int r = 0; r < 8; ++r) Ss[(mt * 16 + mo + r) * 64 + n] = acc[r];
    }
  }
  __syncthreads();
  // ---- softmax (scale folded) + rel-pos bias + shifted-window mask ----
  const float kscale = 0.17677669529663687f;  // 32^-0.5
  if (tid < 49) {
    int i  = tid;
    int ih = i / 7, iw = i % 7;
    int wh = (win % (nh * nh)) / nh, ww = win % nh;
    int idme = 0;
    if (shift) {
      int gh = wh * 7 + ih, gw = ww * 7 + iw;
      int rr = gh < H - 7 ? 0 : (gh < H - shift ? 1 : 2);
      int rc = gw < H - 7 ? 0 : (gw < H - shift ? 1 : 2);
      idme = rr * 3 + rc;
    }
    float mx = -1e30f;
    for (int j = 0; j < 49; ++j) {
      int jh = j / 7, jw = j % 7;
      int rel = (ih - jh + 6) * 13 + (iw - jw + 6);
      float v = Ss[i * 64 + j] * kscale + rpb[rel * nH + h];
      if (shift) {
        int gh2 = wh * 7 + jh, gw2 = ww * 7 + jw;
        int rr2 = gh2 < H - 7 ? 0 : (gh2 < H - shift ? 1 : 2);
        int rc2 = gw2 < H - 7 ? 0 : (gw2 < H - shift ? 1 : 2);
        if (rr2 * 3 + rc2 != idme) v -= 100.f;
      }
      Ss[i * 64 + j] = v;
      mx = fmaxf(mx, v);
    }
    float s = 0.f;
    for (int j = 0; j < 49; ++j) {
      float e = __expf(Ss[i * 64 + j] - mx);
      Ss[i * 64 + j] = e;
      s += e;
    }
    float inv = 1.f / s;
    for (int j = 0; j < 49; ++j) Ss[i * 64 + j] *= inv;
    for (int j = 49; j < 64; ++j) Ss[i * 64 + j] = 0.f;
  } else if (tid < 64) {
    for (int j = 0; j < 64; ++j) Ss[tid * 64 + j] = 0.f;
  }
  __syncthreads();
  // ---- O = P @ V : wave owns m-tile, 2 n-tiles (d=0..31), K=64 ----
  {
    int mt = wave;
    for (int nt2 = 0; nt2 < 2; ++nt2) {
      v8f acc = {};
      for (int k0 = 0; k0 < 64; k0 += 32) {
        v16h a, b;
        int am = mt * 16 + (lane & 15);
        int kb = (lane >> 4) * 8;
#pragma unroll
        for (int i2 = 0; i2 < 8; ++i2) {
          a[i2]     = (_Float16)Ss[am * 64 + k0 + kb + i2];
          a[8 + i2] = (_Float16)Ss[am * 64 + k0 + 16 + kb + i2];
        }
#pragma unroll
        for (int i2 = 0; i2 < 8; ++i2) {
          b[i2]     = vs[(k0 + kb + i2) * 32 + nt2 * 16 + (lane & 15)];
          b[8 + i2] = vs[(k0 + 16 + kb + i2) * 32 + nt2 * 16 + (lane & 15)];
        }
        acc = WMMA_F32_F16(a, b, acc);
      }
      int n  = nt2 * 16 + (lane & 15);
      int mo = (lane >> 4) * 8;
#pragma unroll
      for (int r = 0; r < 8; ++r) {
        int m = mt * 16 + mo + r;
        if (m < 49) out[((size_t)win * 49 + m) * C + h * 32 + n] = (_Float16)acc[r];
      }
    }
  }
}

// ---------------------------------------------------------------------------
// Patch merging: gather 2x2 quadrants -> 4C vector + LayerNorm -> f16
// ---------------------------------------------------------------------------
__global__ void merge_ln_kernel(const float* __restrict__ x, const float* __restrict__ w,
                                const float* __restrict__ b, _Float16* __restrict__ out,
                                int Bn, int H, int C) {
  int H2 = H / 2;
  int row  = (int)(((size_t)blockIdx.x * blockDim.x + threadIdx.x) >> 5);
  int lane = threadIdx.x & 31;
  int total = Bn * H2 * H2;
  if (row >= total) return;
  int ow = row % H2;
  int t  = row / H2;
  int oh = t % H2;
  int bb = t / H2;
  int C4 = 4 * C;
  auto getv = [&](int cc) -> float {
    int q = cc / C, c = cc % C;
    int dh = q & 1, dw = q >> 1;  // q: 0=(0,0) 1=(1,0) 2=(0,1) 3=(1,1)
    return x[(((size_t)bb * H + 2 * oh + dh) * H + 2 * ow + dw) * C + c];
  };
  float s = 0.f;
  for (int cc = lane; cc < C4; cc += 32) s += getv(cc);
  float mu = wave_sum(s) / (float)C4;
  float v = 0.f;
  for (int cc = lane; cc < C4; cc += 32) { float d = getv(cc) - mu; v += d * d; }
  float inv = rsqrtf(wave_sum(v) / (float)C4 + 1e-5f);
  _Float16* o = out + (size_t)row * C4;
  for (int cc = lane; cc < C4; cc += 32) o[cc] = (_Float16)((getv(cc) - mu) * inv * w[cc] + b[cc]);
}

// ---------------------------------------------------------------------------
// Mean over T tokens: in [Bn*T, C] f32 -> out [Bn, C] f32
// ---------------------------------------------------------------------------
__global__ void mean_kernel(const float* __restrict__ in, float* __restrict__ out,
                            int Bn, int T, int C) {
  int idx = blockIdx.x * blockDim.x + threadIdx.x;
  if (idx >= Bn * C) return;
  int c = idx % C, b = idx / C;
  float s = 0.f;
  for (int t = 0; t < T; ++t) s += in[((size_t)b * T + t) * C + c];
  out[idx] = s / (float)T;
}

// ---------------------------------------------------------------------------
// Host side
// ---------------------------------------------------------------------------
static void launch_gemm(const _Float16* A, const float* W, _Float16* Wth,
                        const float* bias, const float* resid,
                        float* Cf, _Float16* Ch, int M, int N, int K, int act,
                        hipStream_t stream) {
  // convert/transpose weight, then GEMM
  size_t wn = (size_t)N * K;
  wt_convert_kernel<<<(int)((wn + 255) / 256), 256, 0, stream>>>(W, Wth, N, K);
  long tiles = (long)(M / 32) * (N / 32);
  int blocks = (int)((tiles + 7) / 8);
  gemm_wmma_kernel<<<blocks, 256, 0, stream>>>(A, Wth, bias, resid, Cf, Ch, M, N, K, act);
}

extern "C" void kernel_launch(void* const* d_in, const int* in_sizes, int n_in,
                              void* d_out, int out_size, void* d_ws, size_t ws_size,
                              hipStream_t stream) {
  (void)out_size; (void)ws_size;
  const int depths[4] = {2, 2, 6, 2};
  const int heads[4]  = {3, 6, 12, 24};

  // Locate x (unique size 64*3*224*224) vs. params (tree_leaves order:
  // sorted dict keys at each level, list order preserved).
  const size_t XSZ = 64ull * 3 * 224 * 224;
  const float* X_IN;
  int pi;
  if ((size_t)in_sizes[0] == XSZ) { X_IN = (const float*)d_in[0]; pi = 1; }
  else                            { X_IN = (const float*)d_in[n_in - 1]; pi = 0; }
  auto next = [&]() -> const float* { return (const float*)d_in[pi++]; };

  const float* fn_b = next();    const float* fn_w = next();
  const float* patch_b = next(); const float* patch_w = next();
  const float* pn_b = next();    const float* pn_w = next();

  struct Blk { const float *fc1_b,*fc1_w,*fc2_b,*fc2_w,*ln1_b,*ln1_w,*ln2_b,*ln2_w,
                           *proj_b,*proj_w,*qkv_b,*qkv_w,*rpb; };
  struct Mrg { const float *norm_b,*norm_w,*red_w; };
  Blk blocks[4][6];
  Mrg merges[3];
  for (int s = 0; s < 4; ++s) {
    for (int d = 0; d < depths[s]; ++d) {
      Blk& bk = blocks[s][d];
      bk.fc1_b = next();  bk.fc1_w = next();
      bk.fc2_b = next();  bk.fc2_w = next();
      bk.ln1_b = next();  bk.ln1_w = next();
      bk.ln2_b = next();  bk.ln2_w = next();
      bk.proj_b = next(); bk.proj_w = next();
      bk.qkv_b = next();  bk.qkv_w = next();
      bk.rpb = next();
    }
    if (s < 3) { merges[s].norm_b = next(); merges[s].norm_w = next(); merges[s].red_w = next(); }
  }

  // Workspace carve. NX = stage-1 activation element count (19,267,584).
  const size_t NX = 64ull * 56 * 56 * 96;
  char* wsb = (char*)d_ws;
  float*    X   = (float*)wsb;     wsb += NX * 4;   // activations (B,H,H,C) f32
  float*    T1  = (float*)wsb;     wsb += NX * 4;   // proj out / final LN (f32)
  float*    T3  = (float*)wsb;     wsb += NX * 4;   // block output (ping-pong with X)
  _Float16* T1h = (_Float16*)wsb;  wsb += NX * 2;   // LN/window/merge f16 out; attn out
  _Float16* T2h = (_Float16*)wsb;  wsb += NX * 8;   // qkv (3C) / fc1 (4C) f16
  _Float16* Wth = (_Float16*)wsb;                   // weight staging (max 3072*768 f16)

  // Patch embedding + patch norm (f32, in place)
  patch_embed_kernel<<<(int)((NX + 255) / 256), 256, 0, stream>>>(X_IN, patch_w, patch_b, X);
  {
    int M = 64 * 56 * 56;
    ln_kernel<float><<<(M * 32 + 255) / 256, 256, 0, stream>>>(X, pn_w, pn_b, X, M, 96);
  }

  int H = 56, C = 96;
  for (int s = 0; s < 4; ++s) {
    int nH = heads[s];
    int nh = H / 7;
    int nWin = nh * nh;
    int Mw = 64 * nWin * 49;  // == 64*H*H
    for (int d = 0; d < depths[s]; ++d) {
      const Blk& bk = blocks[s][d];
      int shift = (d % 2 == 1 && H > 7) ? 3 : 0;
      // LN1 + roll + window partition: X -> T1h [Mw, C] f16
      ln_window_kernel<<<(Mw * 32 + 255) / 256, 256, 0, stream>>>(
          X, bk.ln1_w, bk.ln1_b, T1h, 64, H, C, shift);
      // qkv: T1h @ qkv_w^T + qkv_b -> T2h [Mw, 3C] f16
      launch_gemm(T1h, bk.qkv_w, Wth, bk.qkv_b, nullptr, nullptr, T2h,
                  Mw, 3 * C, C, 0, stream);
      // attention: T2h -> T1h [Mw, C] f16 (T1h free after qkv GEMM)
      attn_kernel<<<64 * nWin * nH, 128, 0, stream>>>(T2h, bk.rpb, T1h, nH, C, nh, shift, H);
      // proj: T1h @ proj_w^T + proj_b -> T1 [Mw, C] f32
      launch_gemm(T1h, bk.proj_w, Wth, bk.proj_b, nullptr, T1, nullptr,
                  Mw, C, C, 0, stream);
      // un-window + roll back + residual into X
      unwindow_add_kernel<<<(int)(((size_t)Mw * C + 255) / 256), 256, 0, stream>>>(
          X, T1, 64, H, C, shift);
      // LN2: X -> T1h f16
      ln_kernel<_Float16><<<(Mw * 32 + 255) / 256, 256, 0, stream>>>(
          X, bk.ln2_w, bk.ln2_b, T1h, Mw, C);
      // fc1 + GELU: T1h -> T2h [Mw, 4C] f16
      launch_gemm(T1h, bk.fc1_w, Wth, bk.fc1_b, nullptr, nullptr, T2h,
                  Mw, 4 * C, C, 1, stream);
      // fc2 + residual(X): T2h -> T3 [Mw, C] f32; ping-pong
      launch_gemm(T2h, bk.fc2_w, Wth, bk.fc2_b, X, T3, nullptr,
                  Mw, C, 4 * C, 0, stream);
      float* tmp = X; X = T3; T3 = tmp;
    }
    if (s < 3) {
      int M2 = 64 * (H / 2) * (H / 2);
      merge_ln_kernel<<<(M2 * 32 + 255) / 256, 256, 0, stream>>>(
          X, merges[s].norm_w, merges[s].norm_b, T1h, 64, H, C);
      launch_gemm(T1h, merges[s].red_w, Wth, nullptr, nullptr, T3, nullptr,
                  M2, 2 * C, 4 * C, 0, stream);
      float* tmp = X; X = T3; T3 = tmp;
      H /= 2; C *= 2;
    }
  }

  // Final LN + spatial mean -> d_out (64, 768)
  ln_kernel<float><<<(64 * 49 * 32 + 255) / 256, 256, 0, stream>>>(
      X, fn_w, fn_b, T1, 64 * 49, 768);
  mean_kernel<<<(64 * 768 + 255) / 256, 256, 0, stream>>>(T1, (float*)d_out, 64, 49, 768);
}
